// Attention_15023795602140
// MI455X (gfx1250) — matrix operand order repaired and checked by reading, not compiled, not run on hardware
//
#include <hip/hip_runtime.h>

typedef _Float16 f16;
typedef _Float16 v16h __attribute__((ext_vector_type(16)));
typedef _Float16 v8h  __attribute__((ext_vector_type(8)));
typedef float    v8f  __attribute__((ext_vector_type(8)));

#define WMMA16(a, b, c) \
  __builtin_amdgcn_wmma_f32_16x16x32_f16(false, (a), false, (b), (short)0, (c), false, false)

// ---------------------------------------------------------------------------
// fp32 -> fp16 elementwise convert (grid-stride)
// ---------------------------------------------------------------------------
__global__ __launch_bounds__(256)
void cvt_f32_f16_kernel(const float* __restrict__ in, f16* __restrict__ out, int n) {
  int i = blockIdx.x * blockDim.x + threadIdx.x;
  int stride = gridDim.x * blockDim.x;
  for (; i < n; i += stride) out[i] = (f16)in[i];
}

// ---------------------------------------------------------------------------
// GEMM, K = N = 1024, M = 4096:  C[m][n] = sum_d A[m][d]*W[n][d] + bias[n]
// Wave tile 32(M) x 64(N): 2 A fragments, 4 B fragments, 8 WMMAs per k-step.
// Block = 128 threads (4 waves) -> block tile 128 x 64.
// mode 0: f16 row-major out; mode 1: f16 transposed per batch [b][n][t];
// mode 2: fp32 + ReLU row-major out.
// ---------------------------------------------------------------------------
__global__ __launch_bounds__(128)
void gemm1024_kernel(const f16* __restrict__ A, const f16* __restrict__ W,
                     const float* __restrict__ bias, void* __restrict__ outp,
                     int mode) {
  const int lane = threadIdx.x & 31;
  const int wave = threadIdx.x >> 5;
  const int lm = lane & 15;       // 0..15
  const int hi = lane >> 4;       // 0 or 1
  const int m0 = blockIdx.x * 128 + wave * 32;
  const int n0 = blockIdx.y * 64;

  v8f acc[2][4];
#pragma unroll
  for (int mt = 0; mt < 2; ++mt)
#pragma unroll
    for (int nt = 0; nt < 4; ++nt) acc[mt][nt] = (v8f){};

  const f16* arow0 = A + (size_t)(m0 + lm) * 1024;
  const f16* arow1 = A + (size_t)(m0 + 16 + lm) * 1024;
  const f16* wr0 = W + (size_t)(n0 + 0 * 16 + lm) * 1024 + hi * 16;
  const f16* wr1 = W + (size_t)(n0 + 1 * 16 + lm) * 1024 + hi * 16;
  const f16* wr2 = W + (size_t)(n0 + 2 * 16 + lm) * 1024 + hi * 16;
  const f16* wr3 = W + (size_t)(n0 + 3 * 16 + lm) * 1024 + hi * 16;

  for (int k0 = 0; k0 < 1024; k0 += 32) {
    __builtin_prefetch(arow0 + k0 + 128, 0, 0);
    __builtin_prefetch(wr0 + k0 + 128, 0, 0);
    // A fragments: row m; elems [0..15] = K k0+16*hi..k0+16*hi+15
    v8h a0lo = *(const v8h*)(arow0 + k0 + hi * 16);
    v8h a0hi = *(const v8h*)(arow0 + k0 + hi * 16 + 8);
    v8h a1lo = *(const v8h*)(arow1 + k0 + hi * 16);
    v8h a1hi = *(const v8h*)(arow1 + k0 + hi * 16 + 8);
    v16h a0, a1;
#pragma unroll
    for (int e = 0; e < 8; ++e) {
      a0[e] = a0lo[e]; a0[e + 8] = a0hi[e];
      a1[e] = a1lo[e]; a1[e + 8] = a1hi[e];
    }
    // B fragments: col n = n0+nt*16+lm; K = k0 + 16*hi + (0..15)
    v16h b0 = *(const v16h*)(wr0 + k0);
    v16h b1 = *(const v16h*)(wr1 + k0);
    v16h b2 = *(const v16h*)(wr2 + k0);
    v16h b3 = *(const v16h*)(wr3 + k0);
    acc[0][0] = WMMA16(a0, b0, acc[0][0]);
    acc[0][1] = WMMA16(a0, b1, acc[0][1]);
    acc[0][2] = WMMA16(a0, b2, acc[0][2]);
    acc[0][3] = WMMA16(a0, b3, acc[0][3]);
    acc[1][0] = WMMA16(a1, b0, acc[1][0]);
    acc[1][1] = WMMA16(a1, b1, acc[1][1]);
    acc[1][2] = WMMA16(a1, b2, acc[1][2]);
    acc[1][3] = WMMA16(a1, b3, acc[1][3]);
  }

  float bv[4];
#pragma unroll
  for (int nt = 0; nt < 4; ++nt) bv[nt] = bias[n0 + nt * 16 + lm];

#pragma unroll
  for (int mt = 0; mt < 2; ++mt) {
#pragma unroll
    for (int r = 0; r < 8; ++r) {
      const int gm = m0 + mt * 16 + r + hi * 8;  // C layout: VGPR r -> r / r+8
#pragma unroll
      for (int nt = 0; nt < 4; ++nt) {
        const int gn = n0 + nt * 16 + lm;
        const float v = acc[mt][nt][r] + bv[nt];
        if (mode == 0) {
          ((f16*)outp)[(size_t)gm * 1024 + gn] = (f16)v;
        } else if (mode == 1) {
          const int bb = gm >> 11, t = gm & 2047;
          ((f16*)outp)[(size_t)bb * 1024 * 2048 + (size_t)gn * 2048 + t] = (f16)v;
        } else {
          ((float*)outp)[(size_t)gm * 1024 + gn] = v > 0.f ? v : 0.f;
        }
      }
    }
  }
}

// ---------------------------------------------------------------------------
// Fused attention per (b, h, 16-query tile), one wave per workgroup.
// LDS: [16][2048] f16 score/prob strip (64 KB) + 2 x [128][72] f16 K-tile
// double buffer (2 x 18 KB), filled with global_load_async_to_lds_b128
// (ASYNCcnt). Async loads complete in order, so after issuing the next
// chunk's 32 transfers, s_wait_asynccnt 32 guarantees the current chunk is
// resident while the next streams in. s_wait_dscnt 0 before re-issuing into
// a buffer closes the LDS WAR race between the wave's DS reads and the async
// DMA writes.
// ---------------------------------------------------------------------------
#define KROW 72  // padded LDS row pitch (f16 elems): 144 B keeps 16B alignment

__device__ __forceinline__ void stage_kchunk(const f16* __restrict__ kx,
                                             f16* __restrict__ dst, int b,
                                             int h, int c, int lane) {
  // 128 rows x 64 dk (128 B/row) = 1024 b128 transfers, 32 per lane
#pragma unroll 4
  for (int j = 0; j < 32; ++j) {
    const int idx = j * 32 + lane;
    const int row = idx >> 3;
    const int seg = idx & 7;
    const f16* g =
        kx + (size_t)(b * 2048 + c * 128 + row) * 1024 + h * 64 + seg * 8;
    f16* l = dst + row * KROW + seg * 8;
    const unsigned lofs = (unsigned)(size_t)l;  // low 32 bits = LDS offset
    const unsigned long long ga = (unsigned long long)g;
    asm volatile("global_load_async_to_lds_b128 %0, %1, off"
                 :: "v"(lofs), "v"(ga) : "memory");
  }
}

__global__ __launch_bounds__(32)
void attn_kernel(const f16* __restrict__ qx, const f16* __restrict__ kx,
                 const f16* __restrict__ vxt, f16* __restrict__ ao,
                 float* __restrict__ attn) {
  extern __shared__ char smem[];
  f16* sh16 = (f16*)smem;                      // [16][2048] scores/probs
  f16* ktile0 = (f16*)(smem + 16 * 2048 * 2);  // [128][KROW] buffer 0
  f16* ktile1 = ktile0 + 128 * KROW;           // [128][KROW] buffer 1

  const int lane = threadIdx.x & 31;
  const int lm = lane & 15;
  const int hi = lane >> 4;
  const int q0 = blockIdx.x * 16;
  const int h = blockIdx.y;
  const int b = blockIdx.z;

  // ---- Q fragments (loaded once): rows q0..q0+15, K = dk 0..31 / 32..63
  const f16* qrow = qx + (size_t)(b * 2048 + q0 + lm) * 1024 + h * 64;
  v16h a0, a1;
  {
    v8h lo = *(const v8h*)(qrow + hi * 16);
    v8h hh = *(const v8h*)(qrow + hi * 16 + 8);
#pragma unroll
    for (int e = 0; e < 8; ++e) { a0[e] = lo[e]; a0[e + 8] = hh[e]; }
  }
  {
    v8h lo = *(const v8h*)(qrow + 32 + hi * 16);
    v8h hh = *(const v8h*)(qrow + 40 + hi * 16);
#pragma unroll
    for (int e = 0; e < 8; ++e) { a1[e] = lo[e]; a1[e + 8] = hh[e]; }
  }

  const float inv_temp = 0.03125f;  // 1/sqrt(1024)

  // ---- Phase 1: scores in 128-key chunks; double-buffered async staging
  stage_kchunk(kx, ktile0, b, h, 0, lane);
  for (int c = 0; c < 16; ++c) {
    const f16* cur = (c & 1) ? ktile1 : ktile0;
    if (c < 15) {
      // WAR fence: prior chunk's ds_loads must have returned before the DMA
      // may overwrite that buffer.
      asm volatile("s_wait_dscnt 0x0" ::: "memory");
      stage_kchunk(kx, (c & 1) ? ktile0 : ktile1, b, h, c + 1, lane);
      asm volatile("s_wait_asynccnt 0x20" ::: "memory");  // chunk c resident
    } else {
      asm volatile("s_wait_asynccnt 0x0" ::: "memory");
    }

    // 8 score tiles of 16 keys each from the staged chunk
#pragma unroll
    for (int t = 0; t < 8; ++t) {
      const f16* kb = cur + (t * 16 + lm) * KROW + hi * 16;
      v16h b0 = *(const v16h*)(kb);        // dk 0..31 slice
      v16h b1 = *(const v16h*)(kb + 32);   // dk 32..63 slice
      v8f cacc = {};
      cacc = WMMA16(a0, b0, cacc);
      cacc = WMMA16(a1, b1, cacc);
#pragma unroll
      for (int r = 0; r < 8; ++r)
        sh16[(r + hi * 8) * 2048 + c * 128 + t * 16 + lm] =
            (f16)(cacc[r] * inv_temp);
    }
  }
  __syncthreads();

  // ---- Phase 2: softmax per query row; probs -> f16 LDS + fp32 global
  for (int r = 0; r < 16; ++r) {
    f16* row = sh16 + r * 2048;
    float mx = -3.0e38f;
    for (int i = lane; i < 2048; i += 32) mx = fmaxf(mx, (float)row[i]);
#pragma unroll
    for (int off = 16; off > 0; off >>= 1) mx = fmaxf(mx, __shfl_xor(mx, off, 32));
    float s = 0.f;
    for (int i = lane; i < 2048; i += 32) {
      const float e = __expf((float)row[i] - mx);
      row[i] = (f16)e;
      s += e;
    }
#pragma unroll
    for (int off = 16; off > 0; off >>= 1) s += __shfl_xor(s, off, 32);
    const float inv = 1.f / s;
    float* arow = attn + ((size_t)((h * 2 + b) * 2048 + q0 + r)) * 2048;
    for (int i = lane; i < 2048; i += 32) {
      const float p = (float)row[i] * inv;
      row[i] = (f16)p;
      arow[i] = p;  // coalesced 128 B per iteration
    }
  }
  __syncthreads();

  // ---- Phase 3: AV. A from LDS (two ds_load_b128), B = vxt rows (contig)
  v8f o0 = {}, o1 = {}, o2 = {}, o3 = {};
  for (int kt = 0; kt < 64; ++kt) {
    const int kb = kt * 32;
    const f16* s1 = sh16 + lm * 2048 + kb + hi * 16;
    v8h alo = *(const v8h*)(s1);
    v8h ahi = *(const v8h*)(s1 + 8);
    v16h a;
#pragma unroll
    for (int e = 0; e < 8; ++e) { a[e] = alo[e]; a[e + 8] = ahi[e]; }
    const f16* vbase =
        vxt + ((size_t)b * 1024 + h * 64 + lm) * 2048 + kb + hi * 16;
    v16h b0 = *(const v16h*)(vbase);
    v16h b1 = *(const v16h*)(vbase + (size_t)16 * 2048);
    v16h b2 = *(const v16h*)(vbase + (size_t)32 * 2048);
    v16h b3 = *(const v16h*)(vbase + (size_t)48 * 2048);
    o0 = WMMA16(a, b0, o0);
    o1 = WMMA16(a, b1, o1);
    o2 = WMMA16(a, b2, o2);
    o3 = WMMA16(a, b3, o3);
  }
#pragma unroll
  for (int r = 0; r < 8; ++r) {
    const size_t gm = (size_t)(b * 2048 + q0 + r + hi * 8);
    ao[gm * 1024 + h * 64 + 0 * 16 + lm] = (f16)o0[r];
    ao[gm * 1024 + h * 64 + 1 * 16 + lm] = (f16)o1[r];
    ao[gm * 1024 + h * 64 + 2 * 16 + lm] = (f16)o2[r];
    ao[gm * 1024 + h * 64 + 3 * 16 + lm] = (f16)o3[r];
  }
}

// ---------------------------------------------------------------------------
// LayerNorm over last dim (1024), one block per token row.
// ---------------------------------------------------------------------------
__global__ __launch_bounds__(256)
void ln_kernel(const float* __restrict__ x, const float* __restrict__ g,
               const float* __restrict__ be, float* __restrict__ out) {
  __shared__ float red[256];
  const int t = threadIdx.x;
  const size_t row = blockIdx.x;
  const float* xr = x + row * 1024;

  float s = 0.f;
  for (int i = t; i < 1024; i += 256) s += xr[i];
  red[t] = s;
  __syncthreads();
  for (int off = 128; off > 0; off >>= 1) {
    if (t < off) red[t] += red[t + off];
    __syncthreads();
  }
  const float mu = red[0] * (1.f / 1024.f);
  __syncthreads();

  float v = 0.f;
  for (int i = t; i < 1024; i += 256) {
    const float d = xr[i] - mu;
    v += d * d;
  }
  red[t] = v;
  __syncthreads();
  for (int off = 128; off > 0; off >>= 1) {
    if (t < off) red[t] += red[t + off];
    __syncthreads();
  }
  const float var = red[0] * (1.f / 1024.f);
  const float rinv = rsqrtf(var + 1e-5f);
  for (int i = t; i < 1024; i += 256)
    out[row * 1024 + i] = (xr[i] - mu) * rinv * g[i] + be[i];
}

// ---------------------------------------------------------------------------
// Host launch
// ---------------------------------------------------------------------------
extern "C" void kernel_launch(void* const* d_in, const int* in_sizes, int n_in,
                              void* d_out, int out_size, void* d_ws,
                              size_t ws_size, hipStream_t stream) {
  const float* k_in = (const float*)d_in[0];
  const float* q_in = (const float*)d_in[1];
  const float* wkw = (const float*)d_in[2];
  const float* wkb = (const float*)d_in[3];
  const float* wqw = (const float*)d_in[4];
  const float* wqb = (const float*)d_in[5];
  const float* wvw = (const float*)d_in[6];
  const float* wvb = (const float*)d_in[7];
  const float* dw = (const float*)d_in[8];
  const float* db = (const float*)d_in[9];
  const float* lng = (const float*)d_in[10];
  const float* lnb = (const float*)d_in[11];

  const int NTOK = 4096;         // B*S
  const int NACT = NTOK * 1024;  // 4,194,304
  const int NW = 1024 * 1024;    // 1,048,576

  char* ws = (char*)d_ws;
  f16* k_h = (f16*)ws;   ws += (size_t)NACT * 2;
  f16* q_h = (f16*)ws;   ws += (size_t)NACT * 2;
  f16* wk_h = (f16*)ws;  ws += (size_t)NW * 2;
  f16* wq_h = (f16*)ws;  ws += (size_t)NW * 2;
  f16* wv_h = (f16*)ws;  ws += (size_t)NW * 2;
  f16* wd_h = (f16*)ws;  ws += (size_t)NW * 2;
  f16* kx_h = (f16*)ws;  ws += (size_t)NACT * 2;
  f16* qx_h = (f16*)ws;  ws += (size_t)NACT * 2;
  f16* vxt_h = (f16*)ws; ws += (size_t)NACT * 2;
  f16* ao_h = (f16*)ws;  ws += (size_t)NACT * 2;
  float* dense_f = (float*)ws;  // NACT fp32

  float* out = (float*)d_out;
  float* attn = out + (size_t)NACT;  // attn output follows `out` flat

  // 1) convert inputs & weights to f16
  cvt_f32_f16_kernel<<<4096, 256, 0, stream>>>(k_in, k_h, NACT);
  cvt_f32_f16_kernel<<<4096, 256, 0, stream>>>(q_in, q_h, NACT);
  cvt_f32_f16_kernel<<<1024, 256, 0, stream>>>(wkw, wk_h, NW);
  cvt_f32_f16_kernel<<<1024, 256, 0, stream>>>(wqw, wq_h, NW);
  cvt_f32_f16_kernel<<<1024, 256, 0, stream>>>(wvw, wv_h, NW);
  cvt_f32_f16_kernel<<<1024, 256, 0, stream>>>(dw, wd_h, NW);

  // 2) projections (v = k in the reference)
  dim3 ggrid(32, 16);
  gemm1024_kernel<<<ggrid, 128, 0, stream>>>(q_h, wq_h, wqb, qx_h, 0);
  gemm1024_kernel<<<ggrid, 128, 0, stream>>>(k_h, wk_h, wkb, kx_h, 0);
  gemm1024_kernel<<<ggrid, 128, 0, stream>>>(k_h, wv_h, wvb, vxt_h, 1);

  // 3) fused attention (double-buffered async K staging)
  const size_t attn_lds =
      16 * 2048 * sizeof(f16) + 2 * 128 * KROW * sizeof(f16);
  attn_kernel<<<dim3(128, 16, 2), 32, attn_lds, stream>>>(qx_h, kx_h, vxt_h,
                                                          ao_h, attn);

  // 4) dense + ReLU (fp32 out)
  gemm1024_kernel<<<ggrid, 128, 0, stream>>>(ao_h, wd_h, db, dense_f, 2);

  // 5) LayerNorm -> final output
  ln_kernel<<<4096, 256, 0, stream>>>(dense_f, lng, lnb, out);
}
